// StreamingConv_52355651338915
// MI455X (gfx1250) — matrix operand order repaired and checked by reading, not compiled, not run on hardware
//
#include <hip/hip_runtime.h>
#include <stdint.h>

typedef __attribute__((ext_vector_type(16))) __bf16 v16bf;
typedef __attribute__((ext_vector_type(8)))  __bf16 v8bf;
typedef __attribute__((ext_vector_type(4)))  __bf16 v4bf;
typedef __attribute__((ext_vector_type(8)))  float  v8f;
typedef __attribute__((ext_vector_type(4)))  float  f32x4;
typedef __attribute__((ext_vector_type(4)))  int    v4i;

static constexpr int MDIM    = 256;    // B (batch)
static constexpr int NDIM    = 1024;   // C_OUT
static constexpr int KDIM    = 16384;  // C_IN * K
static constexpr int KSLICES = 32;
static constexpr int KLEN    = KDIM / KSLICES;  // 512
static constexpr int KSTEP   = 32;
static constexpr int BM      = 128;    // block tile rows (A)
static constexpr int BN      = 128;    // block tile cols (W rows)

// LDS layout (bytes)
static constexpr int F32A    = BM * KSTEP * 4;            // 16384
static constexpr int F32W    = BN * KSTEP * 4;            // 16384
static constexpr int F32BUF  = F32A + F32W;               // 32768
static constexpr int BFA     = BM * KSTEP * 2;            // 8192 (one plane)
static constexpr int BFW     = BN * KSTEP * 2;            // 8192 (one plane)
static constexpr int BFBUF   = 2 * BFA + 2 * BFW;         // 32768 (hi+lo planes)
static constexpr int BFBASE  = 2 * F32BUF;                // 65536
static constexpr int SMEM_SZ = 2 * F32BUF + 2 * BFBUF;    // 131072 (double buffered)

__global__ __launch_bounds__(256)
void sc_zero(float* __restrict__ out, int n) {
  int i = blockIdx.x * 256 + threadIdx.x;
  if (i < n) out[i] = 0.0f;
}

// ---- gfx1250 async global->LDS copy (16B per lane), guarded fallback ----
__device__ __forceinline__ void async_cp16(const float* g, char* l) {
#if __has_builtin(__builtin_amdgcn_global_load_async_to_lds_b128)
  __builtin_amdgcn_global_load_async_to_lds_b128(
      (__attribute__((address_space(1))) v4i*)(g),
      (__attribute__((address_space(3))) v4i*)(l), 0, 0);
#else
  *(f32x4*)l = *(const f32x4*)g;   // fallback: global load + ds store
#endif
}

__device__ __forceinline__ void wait_async_all() {
#if __has_builtin(__builtin_amdgcn_global_load_async_to_lds_b128)
#if __has_builtin(__builtin_amdgcn_s_wait_asynccnt)
  __builtin_amdgcn_s_wait_asynccnt(0);
#else
  asm volatile("s_wait_asynccnt 0" ::: "memory");
#endif
#endif
}

__device__ __forceinline__ v16bf frag16(const char* p0, const char* p1) {
  v8bf a = *(const v8bf*)p0;   // ds_load_b128
  v8bf b = *(const v8bf*)p1;   // ds_load_b128
  return __builtin_shufflevector(a, b, 0, 1, 2, 3, 4, 5, 6, 7,
                                 8, 9, 10, 11, 12, 13, 14, 15);
}

#define WMMA_BF16(c, a, b)                                                     \
  c = __builtin_amdgcn_wmma_f32_16x16x32_bf16(false, a, false, b, (short)0, c, \
                                              false, false)

// D = A * W^T. Grid: 512 = 2 (M/128) x 8 (N/128) x 32 (K slices). 256 thr = 8 waves.
// Per K step: async-stage f32 tiles -> LDS, convert once to bf16 hi/lo planes,
// each wave computes a 64x32 output tile: 24 ds_load_b128 + 24 v_wmma (bf16x3).
__global__ __launch_bounds__(256, 1)
void sc_gemm(const float* __restrict__ A, const float* __restrict__ W,
             float* __restrict__ out) {
  extern __shared__ char smem[];

  const int bid = blockIdx.x;
  const int ks  = bid & (KSLICES - 1);
  const int nb  = (bid >> 5) & 7;
  const int mb  = bid >> 8;

  const int tid  = threadIdx.x;
  const int lane = tid & 31;
  const int wave = tid >> 5;
  const int wm   = wave & 1;    // 2 wave-rows  (64 each)
  const int wn   = wave >> 1;   // 4 wave-cols  (32 each)
  const int r    = lane & 15;
  const int hh   = lane >> 4;

  const int m0     = mb * BM;
  const int n0     = nb * BN;
  const int kBegin = ks * KLEN;

  v8f acc[4][2];
#pragma unroll
  for (int mt = 0; mt < 4; ++mt)
#pragma unroll
    for (int nt = 0; nt < 2; ++nt) acc[mt][nt] = (v8f){};

  // ---- stage: copy A[128 x 32] + W[128 x 32] f32 tiles into LDS (async) ----
  auto stage = [&](int it, int buf) {
    const int kk = kBegin + it * KSTEP;
    char* fA = smem + buf * F32BUF;
    char* fW = fA + F32A;
#pragma unroll
    for (int i = 0; i < 4; ++i) {               // 1024 float4 chunks of A
      int ch = tid + i * 256;
      int row = ch >> 3, c4 = ch & 7;           // 8 float4 per 32-wide row
      async_cp16(A + (size_t)(m0 + row) * KDIM + kk + c4 * 4, fA + ch * 16);
    }
#pragma unroll
    for (int i = 0; i < 4; ++i) {               // 1024 float4 chunks of W
      int ch = tid + i * 256;
      int row = ch >> 3, c4 = ch & 7;
      async_cp16(W + (size_t)(n0 + row) * KDIM + kk + c4 * 4, fW + ch * 16);
    }
  };

  // ---- convert: f32 tile -> bf16 hi/lo planes, once per element per block ----
  auto convertbuf = [&](int buf) {
    const char* f32 = smem + buf * F32BUF;
    char* Ahi = smem + BFBASE + buf * BFBUF;
    char* Alo = Ahi + BFA;
    char* Whi = Alo + BFA;
    char* Wlo = Whi + BFW;
#pragma unroll
    for (int i = 0; i < 8; ++i) {               // 2048 float4 chunks total
      int ch = tid + i * 256;
      f32x4 q = *(const f32x4*)(f32 + ch * 16);
      v4bf h, l;
#pragma unroll
      for (int j = 0; j < 4; ++j) {
        __bf16 hb = (__bf16)q[j];
        h[j] = hb;
        l[j] = (__bf16)(q[j] - (float)hb);
      }
      if (i < 4) {                              // A region: chunks 0..1023
        *(v4bf*)(Ahi + ch * 8) = h;
        *(v4bf*)(Alo + ch * 8) = l;
      } else {                                  // W region: chunks 1024..2047
        *(v4bf*)(Whi + (ch - 1024) * 8) = h;
        *(v4bf*)(Wlo + (ch - 1024) * 8) = l;
      }
    }
  };

  // ---- compute: ds_load fragments in WMMA layout + 24 WMMAs per wave ----
  auto compute = [&](int buf) {
    const char* Ahi = smem + BFBASE + buf * BFBUF;
    const char* Alo = Ahi + BFA;
    const char* Whi = Alo + BFA;
    const char* Wlo = Whi + BFW;
    const int ao = hh * 16;  // A layout: lanes 0-15 K{0..7,16..23}, 16-31 K{8..15,24..31}
    const int bo = hh * 16;  // B layout: lanes 0-15 K{0..7,16..23}, 16-31 K{8..15,24..31}

    v16bf ah[4], al[4];
#pragma unroll
    for (int mt = 0; mt < 4; ++mt) {
      const int ar = (wm * 64 + mt * 16 + r) * 64;   // row stride = 32 bf16 = 64 B
      ah[mt] = frag16(Ahi + ar + ao, Ahi + ar + 32 + ao);
      al[mt] = frag16(Alo + ar + ao, Alo + ar + 32 + ao);
    }
    v16bf bh[2], bl[2];
#pragma unroll
    for (int nt = 0; nt < 2; ++nt) {
      const int br = (wn * 32 + nt * 16 + r) * 64;
      bh[nt] = frag16(Whi + br + bo, Whi + br + bo + 32);
      bl[nt] = frag16(Wlo + br + bo, Wlo + br + bo + 32);
    }
#pragma unroll
    for (int mt = 0; mt < 4; ++mt)
#pragma unroll
      for (int nt = 0; nt < 2; ++nt) {
        WMMA_BF16(acc[mt][nt], ah[mt], bh[nt]);   // hi*hi
        WMMA_BF16(acc[mt][nt], ah[mt], bl[nt]);   // hi*lo
        WMMA_BF16(acc[mt][nt], al[mt], bh[nt]);   // lo*hi
      }
  };

  // ---- pipelined main loop: stage(it+1) overlaps convert+compute(it) ----
  const int NIT = KLEN / KSTEP;  // 16
  stage(0, 0);
  for (int it = 0; it < NIT; ++it) {
    const int cur = it & 1;
    wait_async_all();
    __syncthreads();                       // stage(it) visible to all waves
    if (it + 1 < NIT) stage(it + 1, cur ^ 1);
    convertbuf(cur);
    __syncthreads();                       // bf16 planes visible to all waves
    compute(cur);
  }

  // ---- epilogue: split-K accumulate. C elem i -> M = i + hh*8, N = lane&15 ----
#pragma unroll
  for (int mt = 0; mt < 4; ++mt) {
    const int mA = m0 + wm * 64 + mt * 16 + hh * 8;
#pragma unroll
    for (int nt = 0; nt < 2; ++nt) {
      const int nA = n0 + wn * 32 + nt * 16 + r;
#pragma unroll
      for (int i = 0; i < 8; ++i)
        atomicAdd(out + (size_t)(mA + i) * NDIM + nA, acc[mt][nt][i]);
    }
  }
}

extern "C" void kernel_launch(void* const* d_in, const int* in_sizes, int n_in,
                              void* d_out, int out_size, void* d_ws, size_t ws_size,
                              hipStream_t stream) {
  const float* A = (const float*)d_in[0];  // curr_input [256,1024,16]
  const float* W = (const float*)d_in[1];  // weight     [1024,1024,16]
  float* out = (float*)d_out;              // [256,1024]

  sc_zero<<<dim3((out_size + 255) / 256), dim3(256), 0, stream>>>(out, out_size);

  const int nBlocks = (MDIM / BM) * (NDIM / BN) * KSLICES;  // 512
  sc_gemm<<<dim3(nBlocks), dim3(256), SMEM_SZ, stream>>>(A, W, out);
}
